// FixedAdvancedAttention_59322088292999
// MI455X (gfx1250) — matrix operand order repaired and checked by reading, not compiled, not run on hardware
//
#include <hip/hip_runtime.h>
#include <hip/hip_bf16.h>

// ---------------------------------------------------------------------------
// Types for CDNA5 WMMA (wave32)
// ---------------------------------------------------------------------------
typedef __attribute__((ext_vector_type(16))) __bf16 v16bf;
typedef __attribute__((ext_vector_type(8)))  __bf16 v8bf;
typedef __attribute__((ext_vector_type(8)))  float  v8f;

#define BM 128
#define BN 256
#define BK 32
#define LDST 40                       // LDS row stride in halves (32 + 8 pad, 16B aligned)
#define BUFSZ ((BM + BN) * LDST)      // one double-buffer stage, in halves

enum { EPI_BF16 = 0, EPI_RELU_BF16 = 1, EPI_ADDX_F32 = 2, EPI_BOTH = 3, EPI_FINAL = 4 };

union FragU { v16bf v; v8bf h[2]; };

// ---------------------------------------------------------------------------
// Tiled bf16 WMMA GEMM:  C[M,N] = A[M,K] * W[N,K]^T + bias  (+ fused epilogue)
// A, W row-major, leading dim K (PyTorch Linear layout). K is compile-time.
// Block: 256 threads = 8 waves; block tile 128x256, K-step 32, LDS double-buffered.
// Wave grid 2(M) x 4(N); each wave: 4x4 fragments of 16x16 (64x64 per wave).
// Copy of the next tile is split in two 3-chunk halves interleaved with the
// two 8-WMMA halves so staged registers never exceed 12 VGPRs (no spills
// under the 2-waves/SIMD ~256-VGPR budget).
// ---------------------------------------------------------------------------
template<int EPI, int KV>
__launch_bounds__(256, 2)
__global__ void gemm_bf16_wmma(const __bf16* __restrict__ A,
                               const __bf16* __restrict__ W,
                               const float*  __restrict__ bias,
                               float*        __restrict__ Cf,
                               __bf16*       __restrict__ Cb,
                               const float*  __restrict__ X1,
                               const float*  __restrict__ X2,
                               int M, int N)
{
    __shared__ __bf16 sm[2 * BUFSZ];  // [buf][ A: BM*LDST | B: BN*LDST ]

    const int t    = threadIdx.x;
    const int lane = t & 31;
    const int wave = t >> 5;
    const int wm   = wave & 1;    // 0..1  (M direction)
    const int wn   = wave >> 1;   // 0..3  (N direction)
    const int lh   = lane & 15;   // 0..15
    const int hi   = lane >> 4;   // 0 or 1

    const size_t rowA = (size_t)blockIdx.y * BM;
    const size_t rowB = (size_t)blockIdx.x * BN;

    // tile copy mapping: chunks of 8 halves (16B). A: 512 chunks (2/thread),
    // B: 1024 chunks (4/thread). row = chunk>>2, col = (chunk&3)*8.
    const int r0 = t >> 2;            // 0..63
    const int c0 = (t & 3) * 8;       // 0,8,16,24

    // one per-lane base pointer each; row-group offsets (i*64*KV) are constants
    const __bf16* gA = A + (rowA + r0) * (size_t)KV + c0;
    const __bf16* gB = W + (rowB + r0) * (size_t)KV + c0;

    v8f acc[4][4];
    #pragma unroll
    for (int i = 0; i < 4; ++i)
        #pragma unroll
        for (int j = 0; j < 4; ++j)
            #pragma unroll
            for (int e = 0; e < 8; ++e) acc[i][j][e] = 0.0f;

    const int nK = KV >> 5;

    // prologue: stage tile 0 into buffer 0
    {
        __bf16* dA = sm;
        __bf16* dB = sm + BM * LDST;
        uint4 a0 = *(const uint4*)(gA);
        uint4 a1 = *(const uint4*)(gA + (size_t)64 * KV);
        uint4 b0 = *(const uint4*)(gB);
        uint4 b1 = *(const uint4*)(gB + (size_t)64 * KV);
        uint4 b2 = *(const uint4*)(gB + (size_t)128 * KV);
        uint4 b3 = *(const uint4*)(gB + (size_t)192 * KV);
        *(uint4*)(dA + r0 * LDST + c0)         = a0;
        *(uint4*)(dA + (r0 + 64) * LDST + c0)  = a1;
        *(uint4*)(dB + r0 * LDST + c0)         = b0;
        *(uint4*)(dB + (r0 + 64) * LDST + c0)  = b1;
        *(uint4*)(dB + (r0 + 128) * LDST + c0) = b2;
        *(uint4*)(dB + (r0 + 192) * LDST + c0) = b3;
    }

    for (int kt = 0; kt < nK; ++kt) {
        __syncthreads();                          // buf (kt&1) visible to all waves
        const int  nxt     = kt + 1;
        const bool hasNext = nxt < nK;
        const size_t ko    = (size_t)nxt * BK;

        const __bf16* sA = sm + (kt & 1) * BUFSZ;
        const __bf16* sB = sA + BM * LDST;
        __bf16* dA = sm + (nxt & 1) * BUFSZ;
        __bf16* dB = dA + BM * LDST;

        // --- B fragments for this stage (kept live; 32 VGPRs) ---
        // B 32x16: lanes 0-15 -> K 0..7 & 16..23 ; lanes 16-31 -> K 8..15 & 24..31 (col = lane%16)
        FragU bfr[4];
        #pragma unroll
        for (int nf = 0; nf < 4; ++nf) {
            const __bf16* p = &sB[(wn * 64 + nf * 16 + lh) * LDST + hi * 8];
            bfr[nf].h[0] = *(const v8bf*)p;
            bfr[nf].h[1] = *(const v8bf*)(p + 16);
        }

        // --- stage half 1 of next tile (A rows + B group 0): 12 VGPRs ---
        uint4 ra0, ra1, rb0;
        if (hasNext) {
            ra0 = *(const uint4*)(gA + ko);
            ra1 = *(const uint4*)(gA + ko + (size_t)64 * KV);
            rb0 = *(const uint4*)(gB + ko);
        }
        if (kt + 2 < nK) {
            const size_t ko2 = (size_t)(kt + 2) * BK;
            __builtin_prefetch(gA + ko2, 0, 3);
            __builtin_prefetch(gB + ko2, 0, 3);
        }

        // --- WMMA half 1: mf = 0,1 (A frag streamed; 16 VGPRs) ---
        // A 16x32: lanes 0-15 -> K 0..7 & 16..23 ; lanes 16-31 -> K 8..15 & 24..31
        #pragma unroll
        for (int mf = 0; mf < 2; ++mf) {
            FragU af;
            const __bf16* p = &sA[(wm * 64 + mf * 16 + lh) * LDST + hi * 8];
            af.h[0] = *(const v8bf*)p;
            af.h[1] = *(const v8bf*)(p + 16);
            #pragma unroll
            for (int nf = 0; nf < 4; ++nf)
                acc[mf][nf] = __builtin_amdgcn_wmma_f32_16x16x32_bf16(
                    false, af.v, false, bfr[nf].v, (short)0, acc[mf][nf], false, false);
        }

        if (hasNext) {
            *(uint4*)(dA + r0 * LDST + c0)        = ra0;
            *(uint4*)(dA + (r0 + 64) * LDST + c0) = ra1;
            *(uint4*)(dB + r0 * LDST + c0)        = rb0;
        }

        // --- stage half 2 of next tile (B groups 1..3): 12 VGPRs ---
        uint4 rb1, rb2, rb3;
        if (hasNext) {
            rb1 = *(const uint4*)(gB + ko + (size_t)64 * KV);
            rb2 = *(const uint4*)(gB + ko + (size_t)128 * KV);
            rb3 = *(const uint4*)(gB + ko + (size_t)192 * KV);
        }

        // --- WMMA half 2: mf = 2,3 ---
        #pragma unroll
        for (int mf = 2; mf < 4; ++mf) {
            FragU af;
            const __bf16* p = &sA[(wm * 64 + mf * 16 + lh) * LDST + hi * 8];
            af.h[0] = *(const v8bf*)p;
            af.h[1] = *(const v8bf*)(p + 16);
            #pragma unroll
            for (int nf = 0; nf < 4; ++nf)
                acc[mf][nf] = __builtin_amdgcn_wmma_f32_16x16x32_bf16(
                    false, af.v, false, bfr[nf].v, (short)0, acc[mf][nf], false, false);
        }

        if (hasNext) {
            *(uint4*)(dB + (r0 + 64) * LDST + c0)  = rb1;
            *(uint4*)(dB + (r0 + 128) * LDST + c0) = rb2;
            *(uint4*)(dB + (r0 + 192) * LDST + c0) = rb3;
        }
    }

    // --- epilogue; C/D layout: VGPR r -> M = r + 8*hi, N = lane%16 ---
    #pragma unroll
    for (int mf = 0; mf < 4; ++mf) {
        #pragma unroll
        for (int nf = 0; nf < 4; ++nf) {
            const int    gn  = (int)rowB + wn * 64 + nf * 16 + lh;
            const float  bv  = bias[gn];
            const size_t gm0 = rowA + wm * 64 + mf * 16 + hi * 8;
            #pragma unroll
            for (int r = 0; r < 8; ++r) {
                const size_t idx = (gm0 + r) * (size_t)N + gn;
                float v = acc[mf][nf][r] + bv;
                if (EPI == EPI_BF16) {
                    Cb[idx] = (__bf16)v;
                } else if (EPI == EPI_RELU_BF16) {
                    v = v > 0.0f ? v : 0.0f;
                    Cb[idx] = (__bf16)v;
                } else if (EPI == EPI_ADDX_F32) {
                    Cf[idx] = v + X1[idx];
                } else if (EPI == EPI_BOTH) {
                    Cf[idx] = v;
                    Cb[idx] = (__bf16)v;
                } else { // EPI_FINAL: out = acc + bias + fused + x
                    Cf[idx] = v + X1[idx] + X2[idx];
                }
            }
        }
    }
}

// ---------------------------------------------------------------------------
// fp32 -> bf16 conversion (vectorized, grid-stride); n must be multiple of 4
// ---------------------------------------------------------------------------
__global__ void f32_to_bf16_kernel(const float* __restrict__ in,
                                   __bf16* __restrict__ out, long n)
{
    long i = ((long)blockIdx.x * blockDim.x + threadIdx.x) * 4;
    const long stride = (long)gridDim.x * blockDim.x * 4;
    for (; i + 3 < n; i += stride) {
        const float4 f = *(const float4*)(in + i);
        __bf16 o[4] = { (__bf16)f.x, (__bf16)f.y, (__bf16)f.z, (__bf16)f.w };
        *(uint2*)(out + i) = *(const uint2*)o;
    }
}

// ---------------------------------------------------------------------------
// Row LayerNorm of fp32 [B,D], write bf16 into a column slice of a wider
// bf16 matrix (builds the concatenated [B,3D] buffer without a copy).
// ---------------------------------------------------------------------------
__global__ void ln_to_bf16_kernel(const float* __restrict__ y,
                                  const float* __restrict__ g,
                                  const float* __restrict__ b,
                                  __bf16* __restrict__ out,
                                  int D, int ldo, int colOff)
{
    const int row = blockIdx.x;
    const int t   = threadIdx.x;
    const float* yr = y + (size_t)row * D;

    float s = 0.0f, ss = 0.0f;
    for (int i = t; i < D; i += 256) { const float v = yr[i]; s += v; ss += v * v; }

    __shared__ float r1[256], r2[256];
    r1[t] = s; r2[t] = ss;
    __syncthreads();
    for (int off = 128; off > 0; off >>= 1) {
        if (t < off) { r1[t] += r1[t + off]; r2[t] += r2[t + off]; }
        __syncthreads();
    }
    const float mean = r1[0] / (float)D;
    const float var  = r2[0] / (float)D - mean * mean;
    const float rinv = rsqrtf(var + 1e-5f);

    __bf16* orow = out + (size_t)row * ldo + colOff;
    for (int i = t; i < D; i += 256) {
        const float v = (yr[i] - mean) * rinv * g[i] + b[i];
        orow[i] = (__bf16)v;
    }
}

// ---------------------------------------------------------------------------
// Host launcher
// ---------------------------------------------------------------------------
extern "C" void kernel_launch(void* const* d_in, const int* in_sizes, int n_in,
                              void* d_out, int out_size, void* d_ws, size_t ws_size,
                              hipStream_t stream)
{
    (void)in_sizes; (void)n_in; (void)out_size; (void)ws_size;
    const int B = 16384, D = 2048;
    const size_t BD = (size_t)B * D;

    const float* x = (const float*)d_in[0];
    const float* wv[3] = { (const float*)d_in[1], (const float*)d_in[5], (const float*)d_in[9]  };
    const float* bv[3] = { (const float*)d_in[2], (const float*)d_in[6], (const float*)d_in[10] };
    const float* wo[3] = { (const float*)d_in[3], (const float*)d_in[7], (const float*)d_in[11] };
    const float* bo[3] = { (const float*)d_in[4], (const float*)d_in[8], (const float*)d_in[12] };
    const float* lg[3] = { (const float*)d_in[13], (const float*)d_in[15], (const float*)d_in[17] };
    const float* lb[3] = { (const float*)d_in[14], (const float*)d_in[16], (const float*)d_in[18] };
    const float* fw1  = (const float*)d_in[19]; const float* fb1  = (const float*)d_in[20];
    const float* fw2  = (const float*)d_in[21]; const float* fb2  = (const float*)d_in[22];
    const float* ffw1 = (const float*)d_in[23]; const float* ffb1 = (const float*)d_in[24];
    const float* ffw2 = (const float*)d_in[25]; const float* ffb2 = (const float*)d_in[26];
    float* out = (float*)d_out;

    // workspace carve-out
    char* ws = (char*)d_ws;
    size_t off = 0;
    auto alloc = [&](size_t bytes) -> void* {
        void* p = ws + off;
        off += (bytes + 255) & ~(size_t)255;
        return p;
    };
    __bf16* xb     = (__bf16*)alloc(BD * 2);
    __bf16* wvb[3], *wob[3];
    for (int i = 0; i < 3; ++i) { wvb[i] = (__bf16*)alloc((size_t)D * D * 2);
                                  wob[i] = (__bf16*)alloc((size_t)D * D * 2); }
    __bf16* fw1b   = (__bf16*)alloc((size_t)D * 3 * D * 2);
    __bf16* fw2b   = (__bf16*)alloc((size_t)D * D * 2);
    __bf16* ffw1b  = (__bf16*)alloc((size_t)4 * D * D * 2);
    __bf16* ffw2b  = (__bf16*)alloc((size_t)4 * D * D * 2);
    __bf16* vtmp   = (__bf16*)alloc(BD * 2);
    float*  ytmp   = (float*) alloc(BD * 4);
    __bf16* comb   = (__bf16*)alloc(BD * 3 * 2);
    __bf16* hbuf   = (__bf16*)alloc(BD * 2);
    float*  fusedf = (float*) alloc(BD * 4);
    __bf16* fusedb = (__bf16*)alloc(BD * 2);
    __bf16* f1buf  = (__bf16*)alloc(BD * 4 * 2);

    auto conv = [&](const float* src, __bf16* dst, long n) {
        int blocks = (int)((n / 4 + 255) / 256);
        if (blocks > 32768) blocks = 32768;
        f32_to_bf16_kernel<<<blocks, 256, 0, stream>>>(src, dst, n);
    };

    // --- convert activations & weights to bf16 (part of every call) ---
    conv(x, xb, (long)BD);
    for (int i = 0; i < 3; ++i) { conv(wv[i], wvb[i], (long)D * D);
                                  conv(wo[i], wob[i], (long)D * D); }
    conv(fw1,  fw1b,  (long)D * 3 * D);
    conv(fw2,  fw2b,  (long)D * D);
    conv(ffw1, ffw1b, (long)4 * D * D);
    conv(ffw2, ffw2b, (long)4 * D * D);

    const dim3 blk(256);
    auto grid = [](int N, int M) { return dim3((unsigned)(N / BN), (unsigned)(M / BM)); };

    // --- three attention branches -> LN -> combined[:, i*D:(i+1)*D] ---
    for (int i = 0; i < 3; ++i) {
        gemm_bf16_wmma<EPI_BF16, 2048><<<grid(D, B), blk, 0, stream>>>(
            xb, wvb[i], bv[i], nullptr, vtmp, nullptr, nullptr, B, D);
        gemm_bf16_wmma<EPI_ADDX_F32, 2048><<<grid(D, B), blk, 0, stream>>>(
            vtmp, wob[i], bo[i], ytmp, nullptr, x, nullptr, B, D);
        ln_to_bf16_kernel<<<B, blk, 0, stream>>>(ytmp, lg[i], lb[i], comb, D, 3 * D, i * D);
    }

    // --- fusion MLP: h = relu(comb @ fw1^T + fb1); fused = h @ fw2^T + fb2 ---
    gemm_bf16_wmma<EPI_RELU_BF16, 6144><<<grid(D, B), blk, 0, stream>>>(
        comb, fw1b, fb1, nullptr, hbuf, nullptr, nullptr, B, D);
    gemm_bf16_wmma<EPI_BOTH, 2048><<<grid(D, B), blk, 0, stream>>>(
        hbuf, fw2b, fb2, fusedf, fusedb, nullptr, nullptr, B, D);

    // --- FFN: f1 = relu(fused @ ffw1^T + ffb1); out = x + fused + f1 @ ffw2^T + ffb2 ---
    gemm_bf16_wmma<EPI_RELU_BF16, 2048><<<grid(4 * D, B), blk, 0, stream>>>(
        fusedb, ffw1b, ffb1, nullptr, f1buf, nullptr, nullptr, B, 4 * D);
    gemm_bf16_wmma<EPI_FINAL, 8192><<<grid(D, B), blk, 0, stream>>>(
        f1buf, ffw2b, ffb2, out, nullptr, fusedf, x, B, D);
}
